// GNNStep_88356067213529
// MI455X (gfx1250) — matrix operand order repaired and checked by reading, not compiled, not run on hardware
//
#include <hip/hip_runtime.h>
#include <hip/hip_bf16.h>

// ---------------- problem constants (match reference) ----------------
#define NN      65536      // total nodes
#define CIN     128
#define COUT    128
#define EE      1048576    // edges
#define BB      128        // graphs
#define NPER    512        // nodes per graph
#define KK      256        // kept per graph
#define LIN_IN  32768      // K*COUT
#define LOUT    256
#define EPS_BN  1e-5f
#define SPLITK  32                        // K-splits for the linear GEMM
#define KSTEPS  (LIN_IN / 32 / SPLITK)    // 32 WMMA steps per split

// ---------------- workspace layout (float offsets) ----------------
#define OFF_H      ((size_t)0)                        // h = xn@W        [N,128] f32
#define OFF_CONV   (OFF_H      + (size_t)NN*COUT)     // conv -> x1      [N,128] f32
#define OFF_DEG    (OFF_CONV   + (size_t)NN*COUT)     // deg -> dinv     [N]
#define OFF_T      (OFF_DEG    + (size_t)NN)          // t = x1.w_rel    [N]
#define OFF_ATTN   (OFF_T      + (size_t)NN)          // attn -> score   [N]
#define OFF_STATS  (OFF_ATTN   + (size_t)NN)          // 512: sum,sumsq,scale,shift
#define OFF_TSCORE (OFF_STATS  + (size_t)512)         // top scores      [B*K]
#define OFF_PERM   (OFF_TSCORE + (size_t)BB*KK)       // (int) perm      [B*K]
#define OFF_NEWID  (OFF_PERM   + (size_t)BB*KK)       // (int) new_id    [N]
#define OFF_XNH    (OFF_NEWID  + (size_t)NN)          // f16 BN(x)       [N,128]
#define OFF_WGCN   (OFF_XNH    + (size_t)NN*CIN/2)    // f16 packed gcn_w
#define OFF_WLIN   (OFF_WGCN   + (size_t)8192)        // f16 packed lin_w
#define OFF_XPH    (OFF_WLIN   + (size_t)(LOUT/16)*(LIN_IN/32)*512/2) // f16 x_pool

// ---------------- output layout (float offsets) ----------------
#define O_XPOOL ((size_t)0)                           // [B*K,128]
#define O_EI    (O_XPOOL + (size_t)BB*KK*COUT)        // [2,E]
#define O_EA    (O_EI    + (size_t)2*EE)              // [E]
#define O_BATCH (O_EA    + (size_t)EE)                // [B*K]
#define O_LIN   (O_BATCH + (size_t)BB*KK)             // [B,256]

typedef __attribute__((ext_vector_type(16))) _Float16 v16h;
typedef __attribute__((ext_vector_type(8)))  _Float16 v8h;
typedef __attribute__((ext_vector_type(4)))  _Float16 v4h;
typedef __attribute__((ext_vector_type(8)))  float    v8f;

__device__ __forceinline__ float4 ld4(const float* p) {
    return *reinterpret_cast<const float4*>(p);
}
__device__ __forceinline__ void st4(float* p, float4 v) {
    *reinterpret_cast<float4*>(p) = v;
}
__device__ __forceinline__ v16h cat16(v8h lo, v8h hi) {
    return __builtin_shufflevector(lo, hi, 0, 1, 2, 3, 4, 5, 6, 7,
                                   8, 9, 10, 11, 12, 13, 14, 15);
}

// ============ K0: zero BN sums, set deg = 1.0 (self-loop) ============
__global__ void init_kernel(float* deg, float* stats) {
    int i = blockIdx.x * blockDim.x + threadIdx.x;
    if (i < NN) deg[i] = 1.0f;
    if (i < 256) stats[i] = 0.0f;   // sum[128], sumsq[128]
}

// ============ K1: BN statistics (shared-mem staged atomics) ============
__global__ void bn_stats_kernel(const float* __restrict__ x, float* stats) {
    __shared__ float ssum[CIN];
    __shared__ float ssq[CIN];
    int t = threadIdx.x;
    if (t < CIN) { ssum[t] = 0.0f; ssq[t] = 0.0f; }
    __syncthreads();
    size_t total = (size_t)NN * CIN;
    for (size_t i = (size_t)blockIdx.x * blockDim.x + t; i < total;
         i += (size_t)gridDim.x * blockDim.x) {
        float v = x[i];
        int c = (int)(i & (CIN - 1));
        atomicAdd(&ssum[c], v);
        atomicAdd(&ssq[c], v * v);
    }
    __syncthreads();
    if (t < CIN) {
        atomicAdd(&stats[t], ssum[t]);
        atomicAdd(&stats[CIN + t], ssq[t]);
    }
}

// ============ K1b: finalize -> fused affine scale/shift ============
__global__ void bn_finalize_kernel(float* stats, const float* __restrict__ gamma,
                                   const float* __restrict__ beta) {
    int c = threadIdx.x;
    if (c >= CIN) return;
    float mean = stats[c] * (1.0f / NN);
    float var  = stats[CIN + c] * (1.0f / NN) - mean * mean;
    float inv  = rsqrtf(var + EPS_BN);
    float sc   = gamma[c] * inv;
    stats[256 + c] = sc;                 // scale
    stats[384 + c] = beta[c] - mean * sc;// shift
}

// ============ K1c: xnh = f16(BN(x))  [N,128], float4 granularity ============
__global__ void pack_xn_kernel(const float* __restrict__ x,
                               const float* __restrict__ stats,
                               _Float16* __restrict__ xnh) {
    size_t i4 = (size_t)blockIdx.x * blockDim.x + threadIdx.x;
    if (i4 >= (size_t)NN * (CIN / 4)) return;
    int c4 = ((int)(i4 & 31)) * 4;
    const float* sc = stats + 256 + c4;
    const float* sh = stats + 384 + c4;
    float4 v = ld4(x + i4 * 4);
    v4h o;
    o[0] = (_Float16)(v.x * sc[0] + sh[0]);
    o[1] = (_Float16)(v.y * sc[1] + sh[1]);
    o[2] = (_Float16)(v.z * sc[2] + sh[2]);
    o[3] = (_Float16)(v.w * sc[3] + sh[3]);
    *reinterpret_cast<v4h*>(xnh + i4 * 4) = o;
}

// ============ pack row-major f32 [K,N] weights into WMMA B fragments ======
// layout: wpack[((tn*nKsteps + ks)*32 + lane)*16 + j]
//         lane = column (n = tn*16 + lane&15), K = ks*32 + (lane>>4)*16 + j
__global__ void pack_w_kernel(const float* __restrict__ W, _Float16* __restrict__ wp,
                              int nKsteps, int ldn, size_t total) {
    size_t idx = (size_t)blockIdx.x * blockDim.x + threadIdx.x;
    if (idx >= total) return;
    int j    = (int)(idx & 15);
    int lane = (int)((idx >> 4) & 31);
    int ks   = (int)((idx >> 9) % nKsteps);
    int tn   = (int)((idx >> 9) / nKsteps);
    int k = ks * 32 + (lane >> 4) * 16 + j;
    int n = tn * 16 + (lane & 15);
    wp[idx] = (_Float16)(W[(size_t)k * ldn + n]);
}

// ============ K3b: weighted in-degree via atomics ============
__global__ void deg_edges_kernel(const int* __restrict__ dst,
                                 const float* __restrict__ ea, float* deg) {
    int e = blockIdx.x * blockDim.x + threadIdx.x;
    if (e < EE) atomicAdd(&deg[dst[e]], ea[e]);
}

// ============ K3c: deg -> deg^-0.5 in place ============
__global__ void dinv_kernel(float* deg) {
    int i = blockIdx.x * blockDim.x + threadIdx.x;
    if (i < NN) deg[i] = rsqrtf(deg[i]);
}

// ============ K2: h = xnh @ wgcn  — one wave per 16x128 strip ============
// A fragments (full K=128) loaded once, reused across all 8 N tiles.
// 4096 strips, 32 WMMAs per wave.
__global__ void gemm_gcn_kernel(const _Float16* __restrict__ xnh,
                                const _Float16* __restrict__ wpack,
                                float* __restrict__ h) {
    int wave  = threadIdx.x >> 5;
    int lane  = threadIdx.x & 31;
    int strip = blockIdx.x * 4 + wave;             // M tile, 4096 total
    int row16 = lane & 15;
    int half  = lane >> 4;

    const _Float16* arow = xnh + (size_t)(strip * 16 + row16) * CIN;
    v16h afrag[4];
#pragma unroll
    for (int ks = 0; ks < 4; ++ks) {
        int kb = ks * 32 + half * 16;
        afrag[ks] = cat16(*reinterpret_cast<const v8h*>(arow + kb),
                          *reinterpret_cast<const v8h*>(arow + kb + 8));
    }

#pragma unroll
    for (int tn = 0; tn < 8; ++tn) {
        v8f acc = {};
#pragma unroll
        for (int ks = 0; ks < 4; ++ks) {
            v16h b = *reinterpret_cast<const v16h*>(
                wpack + (((size_t)tn * 4 + ks) * 32 + lane) * 16);
            acc = __builtin_amdgcn_wmma_f32_16x16x32_f16(
                false, afrag[ks], false, b, (short)0, acc, false, false);
        }
        int ncol = tn * 16 + row16;
#pragma unroll
        for (int r = 0; r < 8; ++r) {
            int m = strip * 16 + half * 8 + r;
            h[(size_t)m * COUT + ncol] = acc[r];
        }
    }
}

// ============ K4a: conv = dinv^2 * h (self loop term) ============
__global__ void conv_init_kernel(const float* __restrict__ h,
                                 const float* __restrict__ dinv,
                                 float* __restrict__ conv) {
    size_t i4 = (size_t)blockIdx.x * blockDim.x + threadIdx.x; // float4 index
    if (i4 >= (size_t)NN * (COUT / 4)) return;
    int n = (int)(i4 >> 5);
    float dv = dinv[n];
    float w = dv * dv;
    float4 v = ld4(h + i4 * 4);
    v.x *= w; v.y *= w; v.z *= w; v.w *= w;
    st4(conv + i4 * 4, v);
}

// ============ K4b: conv[dst] += dinv[s]*ea*dinv[d] * h[src] ============
__global__ void edge_scatter_kernel(const float* __restrict__ h,
                                    float* __restrict__ conv,
                                    const float* __restrict__ dinv,
                                    const float* __restrict__ ea,
                                    const int* __restrict__ src,
                                    const int* __restrict__ dst) {
    int e = blockIdx.x * 8 + (threadIdx.x >> 5);
    if (e >= EE) return;
    int lane = threadIdx.x & 31;
    int s = src[e], d = dst[e];
    float w = dinv[s] * ea[e] * dinv[d];
    float4 hv = ld4(h + (size_t)s * COUT + lane * 4);
    float* cp = conv + (size_t)d * COUT + lane * 4;
    atomicAdd(cp + 0, w * hv.x);
    atomicAdd(cp + 1, w * hv.y);
    atomicAdd(cp + 2, w * hv.z);
    atomicAdd(cp + 3, w * hv.w);
}

// ============ K5: x1 = relu(conv + bias) in place ============
__global__ void relu_bias_kernel(float* __restrict__ conv,
                                 const float* __restrict__ bias) {
    size_t i4 = (size_t)blockIdx.x * blockDim.x + threadIdx.x;
    if (i4 >= (size_t)NN * (COUT / 4)) return;
    int c4 = ((int)(i4 & 31)) * 4;
    float4 v = ld4(conv + i4 * 4);
    v.x = fmaxf(v.x + bias[c4 + 0], 0.0f);
    v.y = fmaxf(v.y + bias[c4 + 1], 0.0f);
    v.z = fmaxf(v.z + bias[c4 + 2], 0.0f);
    v.w = fmaxf(v.w + bias[c4 + 3], 0.0f);
    st4(conv + i4 * 4, v);
}

// ============ K6a: per-node dots t = x1.w_rel ; attn = b_rel + x1.w_root ===
__global__ void node_dots_kernel(const float* __restrict__ x1,
                                 const float* __restrict__ wrel,
                                 const float* __restrict__ brel,
                                 const float* __restrict__ wroot,
                                 float* __restrict__ t,
                                 float* __restrict__ attn) {
    int node = (blockIdx.x * blockDim.x + threadIdx.x) >> 5;
    int lane = threadIdx.x & 31;
    if (node >= NN) return;
    const float* r = x1 + (size_t)node * COUT + lane * 4;
    float4 v = ld4(r);
    float4 a = ld4(wrel + lane * 4);
    float4 b = ld4(wroot + lane * 4);
    float dr = v.x * a.x + v.y * a.y + v.z * a.z + v.w * a.w;
    float dt = v.x * b.x + v.y * b.y + v.z * b.z + v.w * b.w;
#pragma unroll
    for (int o = 16; o > 0; o >>= 1) {
        dr += __shfl_down(dr, o, 32);
        dt += __shfl_down(dt, o, 32);
    }
    if (lane == 0) {
        t[node] = dr;
        attn[node] = brel[0] + dt;
    }
}

// ============ K6b: attn[dst] += ea * t[src] ============
__global__ void attn_edges_kernel(const int* __restrict__ src,
                                  const int* __restrict__ dst,
                                  const float* __restrict__ ea,
                                  const float* __restrict__ t,
                                  float* __restrict__ attn) {
    int e = blockIdx.x * blockDim.x + threadIdx.x;
    if (e < EE) atomicAdd(&attn[dst[e]], ea[e] * t[src[e]]);
}

// ============ K6c: score = tanh(attn) in place; new_id = -1 ============
__global__ void tanh_newid_kernel(float* __restrict__ attn, int* __restrict__ newid) {
    int i = blockIdx.x * blockDim.x + threadIdx.x;
    if (i < NN) {
        attn[i] = tanhf(attn[i]);
        newid[i] = -1;
    }
}

// ============ K7: per-graph bitonic top-K (descending, ties: low idx) =====
__global__ __launch_bounds__(NPER)
void topk_kernel(const float* __restrict__ score,
                 float* __restrict__ tscore, int* __restrict__ perm,
                 int* __restrict__ newid, float* __restrict__ out_batch) {
    __shared__ float ss[NPER];
    __shared__ int   si[NPER];
    int b = blockIdx.x, tid = threadIdx.x;
    ss[tid] = score[b * NPER + tid];
    si[tid] = tid;
    __syncthreads();
    for (int k = 2; k <= NPER; k <<= 1) {
        for (int j = k >> 1; j > 0; j >>= 1) {
            int ixj = tid ^ j;
            if (ixj > tid) {
                float s1 = ss[tid], s2 = ss[ixj];
                int i1 = si[tid], i2 = si[ixj];
                bool less1 = (s1 < s2) || (s1 == s2 && i1 > i2);
                bool desc = ((tid & k) == 0);
                bool sw = desc ? less1 : !less1;
                if (sw) { ss[tid] = s2; ss[ixj] = s1; si[tid] = i2; si[ixj] = i1; }
            }
            __syncthreads();
        }
    }
    if (tid < KK) {
        int node = b * NPER + si[tid];
        int g = b * KK + tid;
        perm[g] = node;
        tscore[g] = ss[tid];
        newid[node] = g;
        out_batch[g] = (float)b;
    }
}

// ============ K8: x_pool = x1[perm] * top_score (f32 out + f16 shadow) =====
__global__ void xpool_kernel(const float* __restrict__ x1,
                             const int* __restrict__ perm,
                             const float* __restrict__ tscore,
                             float* __restrict__ out,
                             _Float16* __restrict__ xph) {
    size_t i4 = (size_t)blockIdx.x * blockDim.x + threadIdx.x;
    if (i4 >= (size_t)BB * KK * (COUT / 4)) return;
    int node = (int)(i4 >> 5);
    int c4 = ((int)(i4 & 31)) * 4;
    int p = perm[node];
    float s = tscore[node];
    float4 v = ld4(x1 + (size_t)p * COUT + c4);
    v.x *= s; v.y *= s; v.z *= s; v.w *= s;
    st4(out + i4 * 4, v);
    v4h o;
    o[0] = (_Float16)v.x; o[1] = (_Float16)v.y;
    o[2] = (_Float16)v.z; o[3] = (_Float16)v.w;
    *reinterpret_cast<v4h*>(xph + i4 * 4) = o;
}

// ============ K9: filter_adj -> relabeled edges / masked attr ============
__global__ void filter_edges_kernel(const int* __restrict__ src,
                                    const int* __restrict__ dst,
                                    const float* __restrict__ ea,
                                    const int* __restrict__ newid,
                                    float* __restrict__ out_ei,
                                    float* __restrict__ out_ea) {
    int e = blockIdx.x * blockDim.x + threadIdx.x;
    if (e >= EE) return;
    int ns = newid[src[e]];
    int nd = newid[dst[e]];
    bool ok = (ns >= 0) && (nd >= 0);
    out_ei[e]      = (float)(ok ? ns : -1);
    out_ei[EE + e] = (float)(ok ? nd : -1);
    out_ea[e]      = ok ? ea[e] : 0.0f;
}

// ============ K10a: lin_out = bias (split-K accumulator init) ============
__global__ void lin_init_kernel(float* __restrict__ out, const float* __restrict__ bias) {
    int i = blockIdx.x * blockDim.x + threadIdx.x;
    if (i < BB * LOUT) out[i] = bias[i & (LOUT - 1)];
}

// ============ K10b: lin_out += xph @ wlin  (WMMA, split-K, 4 N-tiles/wave) =
// M=128 (8 tiles), N=256 (4 groups of 4 tiles), K=32768 (32 splits x 32 steps)
// A fragment loaded once per step, reused across 4 WMMAs.
__global__ void gemm_lin_kernel(const _Float16* __restrict__ A,   // [128, 32768] f16
                                const _Float16* __restrict__ wpack,
                                float* __restrict__ out) {        // [128, 256]
    int wave  = threadIdx.x >> 5;
    int lane  = threadIdx.x & 31;
    int wg    = blockIdx.x * 4 + wave;      // 1024 waves
    int split = wg & (SPLITK - 1);
    int rest  = wg >> 5;                    // 32 combos
    int tm    = rest >> 2;                  // 0..7
    int tng   = rest & 3;                   // N group: tiles tng*4 .. tng*4+3
    int row16 = lane & 15;
    int half  = lane >> 4;

    const _Float16* arow = A + (size_t)(tm * 16 + row16) * LIN_IN;
    int ks0 = split * KSTEPS;

    // per-N-tile packed-B base pointers (stride between K-steps: 512 halves)
    const _Float16* bbase[4];
#pragma unroll
    for (int i = 0; i < 4; ++i)
        bbase[i] = wpack +
            (((size_t)(tng * 4 + i) * (LIN_IN / 32) + ks0) * 32 + lane) * 16;

    v8f acc0 = {}, acc1 = {}, acc2 = {}, acc3 = {};
#pragma unroll 2
    for (int s = 0; s < KSTEPS; ++s) {
        int kb = (ks0 + s) * 32 + half * 16;
        v16h a = cat16(*reinterpret_cast<const v8h*>(arow + kb),
                       *reinterpret_cast<const v8h*>(arow + kb + 8));
        if (s + 4 < KSTEPS) {  // stream packed weights ahead (global_prefetch_b8)
            __builtin_prefetch(bbase[0] + (size_t)(s + 4) * 512, 0, 1);
            __builtin_prefetch(bbase[2] + (size_t)(s + 4) * 512, 0, 1);
        }
        v16h b0 = *reinterpret_cast<const v16h*>(bbase[0] + (size_t)s * 512);
        v16h b1 = *reinterpret_cast<const v16h*>(bbase[1] + (size_t)s * 512);
        v16h b2 = *reinterpret_cast<const v16h*>(bbase[2] + (size_t)s * 512);
        v16h b3 = *reinterpret_cast<const v16h*>(bbase[3] + (size_t)s * 512);
        acc0 = __builtin_amdgcn_wmma_f32_16x16x32_f16(
            false, a, false, b0, (short)0, acc0, false, false);
        acc1 = __builtin_amdgcn_wmma_f32_16x16x32_f16(
            false, a, false, b1, (short)0, acc1, false, false);
        acc2 = __builtin_amdgcn_wmma_f32_16x16x32_f16(
            false, a, false, b2, (short)0, acc2, false, false);
        acc3 = __builtin_amdgcn_wmma_f32_16x16x32_f16(
            false, a, false, b3, (short)0, acc3, false, false);
    }

#pragma unroll
    for (int i = 0; i < 4; ++i) {
        const v8f* accp = (i == 0) ? &acc0 : (i == 1) ? &acc1 : (i == 2) ? &acc2 : &acc3;
        int ncol = (tng * 4 + i) * 16 + row16;
#pragma unroll
        for (int r = 0; r < 8; ++r) {
            int m = tm * 16 + half * 8 + r;
            atomicAdd(&out[(size_t)m * LOUT + ncol], (*accp)[r]);
        }
    }
}

// ====================================================================
extern "C" void kernel_launch(void* const* d_in, const int* in_sizes, int n_in,
                              void* d_out, int out_size, void* d_ws, size_t ws_size,
                              hipStream_t stream) {
    const float* x        = (const float*)d_in[0];
    const float* ea       = (const float*)d_in[1];
    const float* bn_gamma = (const float*)d_in[2];
    const float* bn_beta  = (const float*)d_in[3];
    const float* gcn_w    = (const float*)d_in[4];
    const float* gcn_b    = (const float*)d_in[5];
    const float* w_rel    = (const float*)d_in[6];
    const float* b_rel    = (const float*)d_in[7];
    const float* w_root   = (const float*)d_in[8];
    const float* lin_w    = (const float*)d_in[9];
    const float* lin_b    = (const float*)d_in[10];
    const int*   eidx     = (const int*)d_in[11];
    const int*   src      = eidx;
    const int*   dst      = eidx + EE;

    float* out = (float*)d_out;
    float* ws  = (float*)d_ws;

    float*     h      = ws + OFF_H;
    float*     conv   = ws + OFF_CONV;      // becomes x1 after K5
    float*     deg    = ws + OFF_DEG;       // becomes dinv after K3c
    float*     t      = ws + OFF_T;
    float*     attn   = ws + OFF_ATTN;      // becomes score after K6c
    float*     stats  = ws + OFF_STATS;
    float*     tscore = ws + OFF_TSCORE;
    int*       perm   = (int*)(ws + OFF_PERM);
    int*       newid  = (int*)(ws + OFF_NEWID);
    _Float16*  xnh    = (_Float16*)(ws + OFF_XNH);
    _Float16*  wgcn   = (_Float16*)(ws + OFF_WGCN);
    _Float16*  wlin   = (_Float16*)(ws + OFF_WLIN);
    _Float16*  xph    = (_Float16*)(ws + OFF_XPH);

    // --- weight packing (independent of everything else) ---
    pack_w_kernel<<<64, 256, 0, stream>>>(gcn_w, wgcn, CIN / 32, COUT,
                                          (size_t)(COUT / 16) * (CIN / 32) * 512);
    pack_w_kernel<<<32768, 256, 0, stream>>>(lin_w, wlin, LIN_IN / 32, LOUT,
                                             (size_t)(LOUT / 16) * (LIN_IN / 32) * 512);

    // --- BatchNorm ---
    init_kernel<<<NN / 256, 256, 0, stream>>>(deg, stats);
    bn_stats_kernel<<<1024, 256, 0, stream>>>(x, stats);
    bn_finalize_kernel<<<1, 128, 0, stream>>>(stats, bn_gamma, bn_beta);
    pack_xn_kernel<<<(NN * (CIN / 4)) / 256, 256, 0, stream>>>(x, stats, xnh);

    // --- weighted degree -> dinv ---
    deg_edges_kernel<<<EE / 256, 256, 0, stream>>>(dst, ea, deg);
    dinv_kernel<<<NN / 256, 256, 0, stream>>>(deg);

    // --- GCN: GEMM + aggregation ---
    gemm_gcn_kernel<<<1024, 128, 0, stream>>>(xnh, wgcn, h);
    conv_init_kernel<<<(NN * (COUT / 4)) / 256, 256, 0, stream>>>(h, deg, conv);
    edge_scatter_kernel<<<EE / 8, 256, 0, stream>>>(h, conv, deg, ea, src, dst);
    relu_bias_kernel<<<(NN * (COUT / 4)) / 256, 256, 0, stream>>>(conv, gcn_b);

    // --- SAG attention score ---
    node_dots_kernel<<<NN / 8, 256, 0, stream>>>(conv, w_rel, b_rel, w_root, t, attn);
    attn_edges_kernel<<<EE / 256, 256, 0, stream>>>(src, dst, ea, t, attn);
    tanh_newid_kernel<<<NN / 256, 256, 0, stream>>>(attn, newid);

    // --- top-K, gather, filter ---
    topk_kernel<<<BB, NPER, 0, stream>>>(attn, tscore, perm, newid, out + O_BATCH);
    xpool_kernel<<<((size_t)BB * KK * (COUT / 4)) / 256, 256, 0, stream>>>(
        conv, perm, tscore, out + O_XPOOL, xph);
    filter_edges_kernel<<<EE / 256, 256, 0, stream>>>(
        src, dst, ea, newid, out + O_EI, out + O_EA);

    // --- Linear (split-K WMMA) ---
    lin_init_kernel<<<(BB * LOUT) / 256, 256, 0, stream>>>(out + O_LIN, lin_b);
    gemm_lin_kernel<<<256, 128, 0, stream>>>(xph, wlin, out + O_LIN);
}
